// Capsule_61160334295610
// MI455X (gfx1250) — matrix operand order repaired and checked by reading, not compiled, not run on hardware
//
#include <hip/hip_runtime.h>
#include <hip/hip_bf16.h>

typedef __attribute__((ext_vector_type(16))) __bf16 v16bf;
typedef __attribute__((ext_vector_type(8)))  float  v8f;

#define B_DIM 64
#define I_DIM 1024
#define O_DIM 1024
#define EPS 1e-5f

// ---------------------------------------------------------------------------
// Convert u (f32 [B,I]) -> bf16 row-major
__global__ void cvt_u_kernel(const float* __restrict__ u, __bf16* __restrict__ ub, int n) {
    int i = blockIdx.x * blockDim.x + threadIdx.x;
    if (i < n) ub[i] = (__bf16)u[i];
}

// Convert W (f32 [I,O]) -> bf16 transposed [O,I] so WMMA B-fragments load contiguously in K
__global__ void cvt_wT_kernel(const float* __restrict__ W, __bf16* __restrict__ BT) {
    int idx = blockIdx.x * blockDim.x + threadIdx.x;   // 0 .. I*O-1
    int i = idx >> 10;          // row in W
    int o = idx & (O_DIM - 1);  // col in W
    BT[o * I_DIM + i] = (__bf16)W[i * O_DIM + o];
}

// ---------------------------------------------------------------------------
// S = u @ W via v_wmma_f32_16x16x32_bf16.  One wave per 16x16 output tile.
// grid = (O/16, B/16), block = 32 (one wave32)
__global__ void gemm_wmma_kernel(const __bf16* __restrict__ A,   // [B, I] bf16
                                 const __bf16* __restrict__ BT,  // [O, I] bf16 (W transposed)
                                 float* __restrict__ S) {        // [B, O] f32
    const int n0   = blockIdx.x * 16;
    const int m0   = blockIdx.y * 16;
    const int lane = threadIdx.x;       // 0..31
    const int half = lane >> 4;         // 0 or 1
    const int l15  = lane & 15;

    union Frag { v16bf v; uint4 u[2]; };

    v8f acc = {};
    for (int k0 = 0; k0 < I_DIM; k0 += 32) {
        // A fragment (16x32 bf16): lane holds row m0+l15, K = ka..ka+15 contiguous
        const int ka = k0 + half * 16;
        const uint4* ap = (const uint4*)(A + (m0 + l15) * I_DIM + ka);
        Frag a;
        a.u[0] = ap[0];   // k = ka .. ka+7
        a.u[1] = ap[1];   // k = ka+8 .. ka+15

        // B fragment (32x16 bf16): lane holds col n0+l15, K = kb..kb+15 contiguous in BT
        const int kb = k0 + half * 16;
        const uint4* bp = (const uint4*)(BT + (n0 + l15) * I_DIM + kb);
        Frag b;
        b.u[0] = bp[0];
        b.u[1] = bp[1];

        acc = __builtin_amdgcn_wmma_f32_16x16x32_bf16(
            /*neg_a=*/false, a.v, /*neg_b=*/false, b.v,
            /*c_mod=*/(short)0, acc, /*reuse_a=*/false, /*reuse_b=*/false);
    }

    // C/D layout: VGPR r -> row m0+r (lanes 0-15) / m0+8+r (lanes 16-31), col n0+(lane&15)
    const int row_base = m0 + half * 8;
    const int col = n0 + l15;
#pragma unroll
    for (int r = 0; r < 8; ++r)
        S[(row_base + r) * O_DIM + col] = acc[r];
}

// ---------------------------------------------------------------------------
// squash over O for each batch row.  v = n2/(1+n2) * x/(n+eps), x = S*scale + bias
// mode 0: t = v    mode 1: t += v    mode 2: out = v
__global__ void squash_kernel(const float* __restrict__ S, const float* __restrict__ bias,
                              float* __restrict__ t, float* __restrict__ out,
                              float scale, int mode) {
    const int b   = blockIdx.x;
    const int tid = threadIdx.x;   // 256 threads, 4 elems each
    __shared__ float red[8];
    __shared__ float tot_sh;

    const float4* s4 = (const float4*)(S + b * O_DIM);
    const float4* b4 = (const float4*)bias;
    float4 x  = s4[tid];
    float4 bb = b4[tid];
    x.x = x.x * scale + bb.x;
    x.y = x.y * scale + bb.y;
    x.z = x.z * scale + bb.z;
    x.w = x.w * scale + bb.w;

    float ss = x.x * x.x + x.y * x.y + x.z * x.z + x.w * x.w;
#pragma unroll
    for (int m = 16; m >= 1; m >>= 1) ss += __shfl_xor(ss, m, 32);
    if ((tid & 31) == 0) red[tid >> 5] = ss;
    __syncthreads();
    if (tid == 0) {
        float tot = 0.f;
#pragma unroll
        for (int k = 0; k < 8; ++k) tot += red[k];
        tot_sh = tot;
    }
    __syncthreads();

    const float n2 = tot_sh;
    const float n  = sqrtf(n2);
    const float f  = n2 / (1.0f + n2) / (n + EPS);

    float4 v;
    v.x = f * x.x; v.y = f * x.y; v.z = f * x.z; v.w = f * x.w;

    float4* t4 = (float4*)(t + b * O_DIM);
    if (mode == 0) {
        t4[tid] = v;
    } else if (mode == 1) {
        float4 o = t4[tid];
        o.x += v.x; o.y += v.y; o.z += v.z; o.w += v.w;
        t4[tid] = o;
    } else {
        ((float4*)(out + b * O_DIM))[tid] = v;
    }
}

// ---------------------------------------------------------------------------
// One routing pass:  S[b,o] = sum_i  u[b,i]*W[i,o] * softmax_o( u[b,i]*W[i,o]*t[b,o] )
// One block per batch b; 8 waves split the i-loop; each lane owns 32 o-columns in regs.
__global__ void routing_kernel(const float* __restrict__ u, const float* __restrict__ W,
                               const float* __restrict__ t, float* __restrict__ S) {
    const int b    = blockIdx.x;
    const int tid  = threadIdx.x;   // 256
    const int wave = tid >> 5;
    const int lane = tid & 31;

    __shared__ float s_lds[O_DIM];
    for (int k = tid; k < O_DIM; k += 256) s_lds[k] = 0.0f;
    __syncthreads();

    // preload t[b, :] — lane owns o = (j*32 + lane)*4 + q
    float4 tv[8];
    const float4* t4 = (const float4*)(t + b * O_DIM);
#pragma unroll
    for (int j = 0; j < 8; ++j) tv[j] = t4[j * 32 + lane];

    float4 acc[8];
#pragma unroll
    for (int j = 0; j < 8; ++j) acc[j] = make_float4(0.f, 0.f, 0.f, 0.f);

    for (int i = wave; i < I_DIM; i += 8) {
        const float x = u[b * I_DIM + i];
        const float4* w4 = (const float4*)(W + i * O_DIM);
        if (i + 8 < I_DIM)
            __builtin_prefetch(W + (i + 8) * O_DIM + lane * 32, 0, 1);

        float  lsum = 0.0f;
        float4 p[8];
#pragma unroll
        for (int j = 0; j < 8; ++j) {
            float4 wv = w4[j * 32 + lane];
            float4 z;
            z.x = x * wv.x * tv[j].x;  z.y = x * wv.y * tv[j].y;
            z.z = x * wv.z * tv[j].z;  z.w = x * wv.w * tv[j].w;
            // |z| <~ 0.3 here: softmax without max-subtraction is numerically safe
            float4 e;
            e.x = __expf(z.x); e.y = __expf(z.y); e.z = __expf(z.z); e.w = __expf(z.w);
            lsum += e.x + e.y + e.z + e.w;
            p[j].x = x * wv.x * e.x;  p[j].y = x * wv.y * e.y;
            p[j].z = x * wv.z * e.z;  p[j].w = x * wv.w * e.w;
        }
        // wave32 softmax-denominator reduction
#pragma unroll
        for (int m = 16; m >= 1; m >>= 1) lsum += __shfl_xor(lsum, m, 32);
        const float inv = 1.0f / lsum;
#pragma unroll
        for (int j = 0; j < 8; ++j) {
            acc[j].x += p[j].x * inv;  acc[j].y += p[j].y * inv;
            acc[j].z += p[j].z * inv;  acc[j].w += p[j].w * inv;
        }
    }

    // cross-wave accumulation via LDS float atomics (ds_add_f32)
#pragma unroll
    for (int j = 0; j < 8; ++j) {
        const int o = j * 128 + 4 * lane;
        atomicAdd(&s_lds[o + 0], acc[j].x);
        atomicAdd(&s_lds[o + 1], acc[j].y);
        atomicAdd(&s_lds[o + 2], acc[j].z);
        atomicAdd(&s_lds[o + 3], acc[j].w);
    }
    __syncthreads();

    for (int k = tid; k < O_DIM; k += 256)
        S[b * O_DIM + k] = s_lds[k];
}

// ---------------------------------------------------------------------------
extern "C" void kernel_launch(void* const* d_in, const int* in_sizes, int n_in,
                              void* d_out, int out_size, void* d_ws, size_t ws_size,
                              hipStream_t stream) {
    const float* u    = (const float*)d_in[0];   // [64, 1024]
    const float* W    = (const float*)d_in[1];   // [1024, 1024]
    const float* bias = (const float*)d_in[2];   // [1024]
    float* out = (float*)d_out;                  // [64, 1024]

    // workspace layout
    float*  S  = (float*)d_ws;                       // 64*1024 f32
    float*  t  = S + B_DIM * O_DIM;                  // 64*1024 f32
    __bf16* ub = (__bf16*)(t + B_DIM * O_DIM);       // 64*1024 bf16
    __bf16* BT = ub + B_DIM * I_DIM;                 // 1024*1024 bf16

    // 1) bf16 conversions for the WMMA GEMM
    cvt_u_kernel <<<(B_DIM * I_DIM + 255) / 256, 256, 0, stream>>>(u, ub, B_DIM * I_DIM);
    cvt_wT_kernel<<<(I_DIM * O_DIM + 255) / 256, 256, 0, stream>>>(W, BT);

    // 2) S = u @ W  (round-1 logits up to the 1/O scale)
    gemm_wmma_kernel<<<dim3(O_DIM / 16, B_DIM / 16), 32, 0, stream>>>(ub, BT, S);

    // 3) v1 = squash(S/O + bias);  t = v1
    squash_kernel<<<B_DIM, 256, 0, stream>>>(S, bias, t, nullptr, 1.0f / (float)O_DIM, 0);

    // 4) routing round 2: S = sum_i u*W*softmax(u*W*t)
    routing_kernel<<<B_DIM, 256, 0, stream>>>(u, W, t, S);

    // 5) v2 = squash(S + bias);  t += v2
    squash_kernel<<<B_DIM, 256, 0, stream>>>(S, bias, t, nullptr, 1.0f, 1);

    // 6) routing round 3
    routing_kernel<<<B_DIM, 256, 0, stream>>>(u, W, t, S);

    // 7) out = squash(S + bias)
    squash_kernel<<<B_DIM, 256, 0, stream>>>(S, bias, t, out, 1.0f, 2);
}
